// GAT_Net_82824149336811
// MI455X (gfx1250) — matrix operand order repaired and checked by reading, not compiled, not run on hardware
//
#include <hip/hip_runtime.h>
#include <hip/hip_bf16.h>

typedef __attribute__((ext_vector_type(16))) _Float16 v16h;
typedef __attribute__((ext_vector_type(8)))  _Float16 v8h;
typedef __attribute__((ext_vector_type(8)))  float    v8f;

#define N_NODES  100000
#define N_EDGES  1600000
#define ET_EDGES 1700000   // + self loops
#define NTILES   6250      // 100000 / 16

// ---------- helpers: order-preserving float<->uint for atomic max ----------
__device__ __forceinline__ unsigned f2ord(float f) {
    unsigned u = __float_as_uint(f);
    return (u & 0x80000000u) ? ~u : (u | 0x80000000u);
}
__device__ __forceinline__ float ord2f(unsigned u) {
    return (u & 0x80000000u) ? __uint_as_float(u & 0x7FFFFFFFu)
                             : __uint_as_float(~u);
}
#define ORD_NEG_INF 0x007FFFFFu   // f2ord(-inf)

// ---------- generic fills ----------
__global__ void k_fill_f32(float* p, float v, int n) {
    int i = blockIdx.x * blockDim.x + threadIdx.x;
    if (i < n) p[i] = v;
}
__global__ void k_fill_u32(unsigned* p, unsigned v, int n) {
    int i = blockIdx.x * blockDim.x + threadIdx.x;
    if (i < n) p[i] = v;
}

// ---------- weight pack: f32 [64,64] -> f16 WMMA-B fragment order ----------
// frag[((nt*2+kb)*32 + lane)*16 + i] = W[(kb*32 + (i>>3)*16 + (lane>>4)*8 + (i&7))*64 + nt*16 + (lane&15)]
__global__ void k_pack_w(const float* __restrict__ W, _Float16* __restrict__ frag) {
    int t = blockIdx.x * blockDim.x + threadIdx.x;
    if (t >= 4096) return;
    int i  = t & 15;
    int l  = (t >> 4) & 31;
    int kb = (t >> 9) & 1;
    int nt = t >> 10;
    int k = kb * 32 + (i >> 3) * 16 + (l >> 4) * 8 + (i & 7);
    int n = nt * 16 + (l & 15);
    frag[t] = (_Float16)W[k * 64 + n];
}

// ---------- embedding: h = (birth + gender + mean(symp)) / 3 ----------
__global__ void k_embed(const int* __restrict__ x,
                        const float* __restrict__ bt,   // [4,64]
                        const float* __restrict__ gt,   // [2,64]
                        const float* __restrict__ st,   // [15,3,64]
                        float* __restrict__ h, _Float16* __restrict__ h16) {
    int t = blockIdx.x * blockDim.x + threadIdx.x;
    if (t >= N_NODES * 64) return;
    int n = t >> 6, d = t & 63;
    const int* xr = x + n * 20;
    int b = 0;
#pragma unroll
    for (int j = 1; j < 4; ++j) if (xr[j] != 0) b = j;   // argmax of one-hot
    int g = xr[4];
    float fe = 0.f;
#pragma unroll
    for (int s = 0; s < 15; ++s) {
        int v = xr[5 + s];
        fe += st[(s * 3 + v) * 64 + d];
    }
    float hv = (bt[b * 64 + d] + gt[g * 64 + d] + fe * (1.f / 15.f)) * (1.f / 3.f);
    h[t] = hv;
    h16[t] = (_Float16)hv;
}

// ---------- WMMA GEMM: xl = A@Wl + bl ; xr = A@Wr + br  (A: [N,64] f16) ----------
// Wl/Wr are pre-packed fragment-order weights (k_pack_w).
__global__ void k_gemm_pair(const _Float16* __restrict__ A,
                            const _Float16* __restrict__ Wl,
                            const _Float16* __restrict__ Wr,
                            const float* __restrict__ bl,
                            const float* __restrict__ br,
                            float* __restrict__ xl,
                            float* __restrict__ xr) {
    int wave = (blockIdx.x * blockDim.x + threadIdx.x) >> 5;
    int lane = threadIdx.x & 31;
    if (wave >= NTILES) return;

    // A fragments (16-bit 16x32 layout): lane&15 = M row,
    // lane<16 holds K {0-7,16-23}+kbase, lane>=16 holds K {8-15,24-31}+kbase
    // Each 8-half run is contiguous -> vector b128 loads.
    const _Float16* Ar = A + (size_t)(wave * 16 + (lane & 15)) * 64;
    int g8 = (lane >> 4) * 8;
    union { v16h v; v8h p[2]; } A0, A1;
    A0.p[0] = *(const v8h*)(Ar + g8);
    A0.p[1] = *(const v8h*)(Ar + 16 + g8);
    A1.p[0] = *(const v8h*)(Ar + 32 + g8);
    A1.p[1] = *(const v8h*)(Ar + 48 + g8);

    int col = lane & 15;

#pragma unroll
    for (int w = 0; w < 2; ++w) {
        const _Float16* Wf = w ? Wr : Wl;
        const float* bias  = w ? br : bl;
        float* out         = w ? xr : xl;
#pragma unroll
        for (int nt = 0; nt < 4; ++nt) {
            // pre-swizzled: lane's 16 halfs contiguous (32 bytes) per (nt,kb)
            const _Float16* F0 = Wf + ((size_t)(nt * 2 + 0) * 32 + lane) * 16;
            const _Float16* F1 = Wf + ((size_t)(nt * 2 + 1) * 32 + lane) * 16;
            union { v16h v; v8h p[2]; } B0, B1;
            B0.p[0] = *(const v8h*)(F0);
            B0.p[1] = *(const v8h*)(F0 + 8);
            B1.p[0] = *(const v8h*)(F1);
            B1.p[1] = *(const v8h*)(F1 + 8);

            v8f c = {};
            c = __builtin_amdgcn_wmma_f32_16x16x32_f16(false, A0.v, false, B0.v,
                                                       (short)0, c, false, false);
            c = __builtin_amdgcn_wmma_f32_16x16x32_f16(false, A1.v, false, B1.v,
                                                       (short)0, c, false, false);
            int n = nt * 16 + col;
            float bv = bias[n];
#pragma unroll
            for (int r = 0; r < 8; ++r) {
                int row = wave * 16 + (lane >> 4) * 8 + r;
                out[(size_t)row * 64 + n] = c[r] + bv;
            }
        }
    }
}

// ---------- layer-1 edge logits (4 heads of 16) : wave per edge ----------
__global__ void k_logits4(const int* __restrict__ ei,
                          const float* __restrict__ xl, const float* __restrict__ xr,
                          const float* __restrict__ att,   // [4*16] flat
                          float* __restrict__ logit, unsigned* __restrict__ m) {
    int wid  = (blockIdx.x * blockDim.x + threadIdx.x) >> 5;
    int lane = threadIdx.x & 31;
    if (wid >= ET_EDGES) return;
    int s, dv;
    if (wid < N_EDGES) { s = ei[wid]; dv = ei[N_EDGES + wid]; }
    else               { s = dv = wid - N_EDGES; }
    const float* L = xl + (size_t)s  * 64;
    const float* R = xr + (size_t)dv * 64;
    int d0 = lane, d1 = lane + 32;
    float v0 = L[d0] + R[d0]; v0 = v0 > 0.f ? v0 : 0.2f * v0;
    float v1 = L[d1] + R[d1]; v1 = v1 > 0.f ? v1 : 0.2f * v1;
    float p0 = v0 * att[d0];
    float p1 = v1 * att[d1];
#pragma unroll
    for (int off = 8; off; off >>= 1) {
        p0 += __shfl_xor(p0, off, 32);
        p1 += __shfl_xor(p1, off, 32);
    }
    if ((lane & 15) == 0) {
        int hh = lane >> 4;                 // 0 or 1
        logit[(size_t)wid * 4 + hh]     = p0;   // heads 0/1
        logit[(size_t)wid * 4 + 2 + hh] = p1;   // heads 2/3
        atomicMax(m + (size_t)dv * 4 + hh,     f2ord(p0));
        atomicMax(m + (size_t)dv * 4 + 2 + hh, f2ord(p1));
    }
}

// ---------- layer-1 exp + denom : thread per (edge,head) ----------
__global__ void k_exp4(const int* __restrict__ ei, float* __restrict__ logit,
                       const unsigned* __restrict__ m, float* __restrict__ den) {
    int t = blockIdx.x * blockDim.x + threadIdx.x;
    if (t >= ET_EDGES * 4) return;
    int e = t >> 2, hh = t & 3;
    int dv = (e < N_EDGES) ? ei[N_EDGES + e] : (e - N_EDGES);
    float a = __expf(logit[t] - ord2f(m[(size_t)dv * 4 + hh]));
    logit[t] = a;
    atomicAdd(den + (size_t)dv * 4 + hh, a);
}

// ---------- layer-1 scatter: acc[dst] += alpha * xl[src] ----------
__global__ void k_scatter4(const int* __restrict__ ei, const float* __restrict__ xl,
                           const float* __restrict__ logit, const float* __restrict__ den,
                           float* __restrict__ acc) {
    int wid  = (blockIdx.x * blockDim.x + threadIdx.x) >> 5;
    int lane = threadIdx.x & 31;
    if (wid >= ET_EDGES) return;
    int s, dv;
    if (wid < N_EDGES) { s = ei[wid]; dv = ei[N_EDGES + wid]; }
    else               { s = dv = wid - N_EDGES; }
    int d0 = lane, d1 = lane + 32;
    float a0 = logit[(size_t)wid * 4 + (d0 >> 4)] / (den[(size_t)dv * 4 + (d0 >> 4)] + 1e-16f);
    float a1 = logit[(size_t)wid * 4 + (d1 >> 4)] / (den[(size_t)dv * 4 + (d1 >> 4)] + 1e-16f);
    atomicAdd(acc + (size_t)dv * 64 + d0, a0 * xl[(size_t)s * 64 + d0]);
    atomicAdd(acc + (size_t)dv * 64 + d1, a1 * xl[(size_t)s * 64 + d1]);
}

// ---------- bias + ELU, produce f32 + f16 for next GEMM ----------
__global__ void k_bias_elu(const float* __restrict__ acc, const float* __restrict__ bias,
                           float* __restrict__ h, _Float16* __restrict__ h16) {
    int t = blockIdx.x * blockDim.x + threadIdx.x;
    if (t >= N_NODES * 64) return;
    int d = t & 63;
    float v = acc[t] + bias[d];
    v = v > 0.f ? v : (__expf(v) - 1.f);
    h[t] = v;
    h16[t] = (_Float16)v;
}

// ---------- layer-2 edge logits (single head of 64) ----------
__global__ void k_logits1(const int* __restrict__ ei,
                          const float* __restrict__ xl, const float* __restrict__ xr,
                          const float* __restrict__ att,   // [64]
                          float* __restrict__ logit, unsigned* __restrict__ m) {
    int wid  = (blockIdx.x * blockDim.x + threadIdx.x) >> 5;
    int lane = threadIdx.x & 31;
    if (wid >= ET_EDGES) return;
    int s, dv;
    if (wid < N_EDGES) { s = ei[wid]; dv = ei[N_EDGES + wid]; }
    else               { s = dv = wid - N_EDGES; }
    const float* L = xl + (size_t)s  * 64;
    const float* R = xr + (size_t)dv * 64;
    int d0 = lane, d1 = lane + 32;
    float v0 = L[d0] + R[d0]; v0 = v0 > 0.f ? v0 : 0.2f * v0;
    float v1 = L[d1] + R[d1]; v1 = v1 > 0.f ? v1 : 0.2f * v1;
    float p = v0 * att[d0] + v1 * att[d1];
#pragma unroll
    for (int off = 16; off; off >>= 1) p += __shfl_xor(p, off, 32);
    if (lane == 0) {
        logit[wid] = p;
        atomicMax(m + dv, f2ord(p));
    }
}

__global__ void k_exp1(const int* __restrict__ ei, float* __restrict__ logit,
                       const unsigned* __restrict__ m, float* __restrict__ den) {
    int e = blockIdx.x * blockDim.x + threadIdx.x;
    if (e >= ET_EDGES) return;
    int dv = (e < N_EDGES) ? ei[N_EDGES + e] : (e - N_EDGES);
    float a = __expf(logit[e] - ord2f(m[dv]));
    logit[e] = a;
    atomicAdd(den + dv, a);
}

__global__ void k_scatter1(const int* __restrict__ ei, const float* __restrict__ xl,
                           const float* __restrict__ logit, const float* __restrict__ den,
                           float* __restrict__ acc) {
    int wid  = (blockIdx.x * blockDim.x + threadIdx.x) >> 5;
    int lane = threadIdx.x & 31;
    if (wid >= ET_EDGES) return;
    int s, dv;
    if (wid < N_EDGES) { s = ei[wid]; dv = ei[N_EDGES + wid]; }
    else               { s = dv = wid - N_EDGES; }
    float alpha = logit[wid] / (den[dv] + 1e-16f);
    int d0 = lane, d1 = lane + 32;
    atomicAdd(acc + (size_t)dv * 64 + d0, alpha * xl[(size_t)s * 64 + d0]);
    atomicAdd(acc + (size_t)dv * 64 + d1, alpha * xl[(size_t)s * 64 + d1]);
}

// ---------- final: out[n] = (acc[n]+bias2) . linW + linb ----------
__global__ void k_out(const float* __restrict__ acc, const float* __restrict__ bias2,
                      const float* __restrict__ linW, const float* __restrict__ linb,
                      float* __restrict__ out) {
    int n    = (blockIdx.x * blockDim.x + threadIdx.x) >> 5;
    int lane = threadIdx.x & 31;
    if (n >= N_NODES) return;
    int d0 = lane, d1 = lane + 32;
    const float* a = acc + (size_t)n * 64;
    float p = (a[d0] + bias2[d0]) * linW[d0] + (a[d1] + bias2[d1]) * linW[d1];
#pragma unroll
    for (int off = 16; off; off >>= 1) p += __shfl_xor(p, off, 32);
    if (lane == 0) out[n] = p + linb[0];
}

extern "C" void kernel_launch(void* const* d_in, const int* in_sizes, int n_in,
                              void* d_out, int out_size, void* d_ws, size_t ws_size,
                              hipStream_t stream) {
    const int*   x    = (const int*)  d_in[0];
    const int*   ei   = (const int*)  d_in[1];
    const float* bt   = (const float*)d_in[2];
    const float* gt   = (const float*)d_in[3];
    const float* st   = (const float*)d_in[4];
    const float* Wl1  = (const float*)d_in[5];
    const float* bl1  = (const float*)d_in[6];
    const float* Wr1  = (const float*)d_in[7];
    const float* br1  = (const float*)d_in[8];
    const float* att1 = (const float*)d_in[9];
    const float* bias1= (const float*)d_in[10];
    const float* Wl2  = (const float*)d_in[11];
    const float* bl2  = (const float*)d_in[12];
    const float* Wr2  = (const float*)d_in[13];
    const float* br2  = (const float*)d_in[14];
    const float* att2 = (const float*)d_in[15];
    const float* bias2= (const float*)d_in[16];
    const float* linW = (const float*)d_in[17];
    const float* linb = (const float*)d_in[18];
    float* out = (float*)d_out;

    // workspace carve-up (256B aligned chunks)
    char* ws = (char*)d_ws;
    size_t off = 0;
    auto carve = [&](size_t bytes) -> char* {
        char* p = ws + off;
        off = (off + bytes + 255) & ~(size_t)255;
        return p;
    };
    float*     h     = (float*)    carve((size_t)N_NODES * 64 * 4);
    _Float16*  h16   = (_Float16*) carve((size_t)N_NODES * 64 * 2);
    float*     xl    = (float*)    carve((size_t)N_NODES * 64 * 4);
    float*     xr    = (float*)    carve((size_t)N_NODES * 64 * 4);
    float*     acc   = (float*)    carve((size_t)N_NODES * 64 * 4);
    float*     logit = (float*)    carve((size_t)ET_EDGES * 4 * 4);
    unsigned*  mbuf  = (unsigned*) carve((size_t)N_NODES * 4 * 4);
    float*     den   = (float*)    carve((size_t)N_NODES * 4 * 4);
    _Float16*  w16   = (_Float16*) carve((size_t)4 * 64 * 64 * 2);
    _Float16* Wl1h = w16;
    _Float16* Wr1h = w16 + 4096;
    _Float16* Wl2h = w16 + 8192;
    _Float16* Wr2h = w16 + 12288;

    const int B = 256;
    const int gEdgeWave = (ET_EDGES + 7) / 8;      // 8 waves / block
    const int gNodeElem = (N_NODES * 64 + B - 1) / B;

    // weights -> f16 pre-swizzled WMMA-B fragments
    k_pack_w<<<16, B, 0, stream>>>(Wl1, Wl1h);
    k_pack_w<<<16, B, 0, stream>>>(Wr1, Wr1h);
    k_pack_w<<<16, B, 0, stream>>>(Wl2, Wl2h);
    k_pack_w<<<16, B, 0, stream>>>(Wr2, Wr2h);

    // embeddings
    k_embed<<<gNodeElem, B, 0, stream>>>(x, bt, gt, st, h, h16);

    // ===== layer 1 =====
    k_fill_f32<<<gNodeElem, B, 0, stream>>>(acc, 0.f, N_NODES * 64);
    k_fill_u32<<<(N_NODES * 4 + B - 1) / B, B, 0, stream>>>(mbuf, ORD_NEG_INF, N_NODES * 4);
    k_fill_f32<<<(N_NODES * 4 + B - 1) / B, B, 0, stream>>>(den, 0.f, N_NODES * 4);

    k_gemm_pair<<<(NTILES + 7) / 8, B, 0, stream>>>(h16, Wl1h, Wr1h, bl1, br1, xl, xr);

    k_logits4<<<gEdgeWave, B, 0, stream>>>(ei, xl, xr, att1, logit, mbuf);
    k_exp4<<<(ET_EDGES * 4 + B - 1) / B, B, 0, stream>>>(ei, logit, mbuf, den);
    k_scatter4<<<gEdgeWave, B, 0, stream>>>(ei, xl, logit, den, acc);

    k_bias_elu<<<gNodeElem, B, 0, stream>>>(acc, bias1, h, h16);

    // ===== layer 2 =====
    k_fill_f32<<<gNodeElem, B, 0, stream>>>(acc, 0.f, N_NODES * 64);
    k_fill_u32<<<(N_NODES + B - 1) / B, B, 0, stream>>>(mbuf, ORD_NEG_INF, N_NODES);
    k_fill_f32<<<(N_NODES + B - 1) / B, B, 0, stream>>>(den, 0.f, N_NODES);

    k_gemm_pair<<<(NTILES + 7) / 8, B, 0, stream>>>(h16, Wl2h, Wr2h, bl2, br2, xl, xr);

    k_logits1<<<gEdgeWave, B, 0, stream>>>(ei, xl, xr, att2, logit, mbuf);
    k_exp1<<<(ET_EDGES + B - 1) / B, B, 0, stream>>>(ei, logit, mbuf, den);
    k_scatter1<<<gEdgeWave, B, 0, stream>>>(ei, xl, logit, den, acc);

    // final linear
    k_out<<<(N_NODES + 7) / 8, B, 0, stream>>>(acc, bias2, linW, linb, out);
}
